// SelfAttentionModule_33432025432140
// MI455X (gfx1250) — matrix operand order repaired and checked by reading, not compiled, not run on hardware
//
#include <hip/hip_runtime.h>

#define B_  8
#define C_  512
#define HW_ 4096
#define CR_ 64

typedef __attribute__((ext_vector_type(16))) __bf16 v16bf;
typedef __attribute__((ext_vector_type(8)))  float  v8f;
typedef __attribute__((ext_vector_type(4)))  unsigned int v4u;
typedef __attribute__((ext_vector_type(8)))  int v8i;
typedef __attribute__((ext_vector_type(4)))  int v4i;

union Frag32 { uint4 q[2]; unsigned short u[16]; v16bf v; };
union Pack8  { unsigned short u[8]; uint4 q; };

__device__ __forceinline__ unsigned short f2bf(float f) {
  unsigned int x = __float_as_uint(f);
  x += 0x7FFFu + ((x >> 16) & 1u);            // round-to-nearest-even
  return (unsigned short)(x >> 16);
}

// A operand (16x32 bf16, MxK): lane = row (l16); element e -> K = (e<8?e:e+8)+hf*8
__device__ __forceinline__ v16bf frag_a(const unsigned short* row, int kb, int hf) {
  Frag32 f;
  f.q[0] = *(const uint4*)(row + kb + hf * 8);
  f.q[1] = *(const uint4*)(row + kb + 16 + hf * 8);
  return f.v;
}

// B operand (32x16 bf16, KxN): lane = col (l16); element e -> K = (e<8?e:e+8)+hf*8
__device__ __forceinline__ v16bf frag_b(const unsigned short* row, int kb, int hf) {
  Frag32 f;
  f.q[0] = *(const uint4*)(row + kb + hf * 8);
  f.q[1] = *(const uint4*)(row + kb + 16 + hf * 8);
  return f.v;
}

// Async global->LDS copy, 16B per lane, tracked by ASYNCcnt (GV mode).
__device__ __forceinline__ void async_b128(unsigned lds_off, unsigned long long gaddr) {
  asm volatile("global_load_async_to_lds_b128 %0, %1, off"
               :: "v"(lds_off), "v"(gaddr) : "memory");
}

// ---------------------------------------------------------------------------
// Pass 1: fused q/k/v projection GEMM (4 WMMAs).
//   Qb,Kb: [B][N][64] bf16   Vc: [B][C][N] bf16
// ---------------------------------------------------------------------------
__global__ __launch_bounds__(256)
void proj_qkv_wmma(const float* __restrict__ x,
                   const float* __restrict__ wq, const float* __restrict__ bq,
                   const float* __restrict__ wk, const float* __restrict__ bk,
                   const float* __restrict__ wv, const float* __restrict__ bv,
                   unsigned short* __restrict__ Qb,
                   unsigned short* __restrict__ Kb,
                   unsigned short* __restrict__ Vc)
{
  __shared__ __align__(16) unsigned short sA[64][72];  // [n][c] bf16
  __shared__ __align__(16) unsigned short sB[64][72];  // [j][c] bf16
  const int b   = blockIdx.z;
  const int n0  = blockIdx.x * 64;
  const int mt  = blockIdx.y;
  const int tid = threadIdx.x;
  const int wave = tid >> 5;
  const int lane = tid & 31;
  const int hf   = lane >> 4;
  const int l16  = lane & 15;

  const float* W; const float* bias;
  if (mt == 0)      { W = wq; bias = bq; }
  else if (mt == 1) { W = wk; bias = bk; }
  else              { W = wv + (size_t)(mt - 2) * 64 * C_; bias = bv + (mt - 2) * 64; }

  const float* xb = x + (size_t)b * C_ * HW_;

  const int rs    = (2 * wave) >> 2;
  const int cs0   = (2 * wave) & 3;
  const int jcol0 = cs0 * 16 + l16;
  const int jcol1 = jcol0 + 16;

  v8f acc0 = {0,0,0,0,0,0,0,0};
  v8f acc1 = {0,0,0,0,0,0,0,0};

  for (int kt = 0; kt < 8; ++kt) {
    const int c0 = kt * 64;
    for (int i = tid; i < 1024; i += 256) {        // A: x[c][n] -> sA[n][c]
      int c = i >> 4, nq = (i & 15) * 4;
      float4 f = *(const float4*)(xb + (size_t)(c0 + c) * HW_ + n0 + nq);
      sA[nq + 0][c] = f2bf(f.x);
      sA[nq + 1][c] = f2bf(f.y);
      sA[nq + 2][c] = f2bf(f.z);
      sA[nq + 3][c] = f2bf(f.w);
    }
    for (int i = tid; i < 1024; i += 256) {        // B: W[j][c] -> sB[j][c]
      int j = i >> 4, cq = (i & 15) * 4;
      float4 f = *(const float4*)(W + (size_t)j * C_ + c0 + cq);
      sB[j][cq + 0] = f2bf(f.x);
      sB[j][cq + 1] = f2bf(f.y);
      sB[j][cq + 2] = f2bf(f.z);
      sB[j][cq + 3] = f2bf(f.w);
    }
    __syncthreads();
    #pragma unroll
    for (int ks = 0; ks < 2; ++ks) {
      const int kb = ks * 32;
      v16bf a  = frag_a(&sA[rs * 16 + l16][0], kb, hf);
      v16bf b0 = frag_b(&sB[jcol0][0], kb, hf);
      v16bf b1 = frag_b(&sB[jcol1][0], kb, hf);
      acc0 = __builtin_amdgcn_wmma_f32_16x16x32_bf16(false, a, false, b0, (short)0, acc0, false, false);
      acc1 = __builtin_amdgcn_wmma_f32_16x16x32_bf16(false, a, false, b1, (short)0, acc1, false, false);
    }
    __syncthreads();
  }

  const float bias0 = bias[jcol0];
  const float bias1 = bias[jcol1];
  const int nbase = n0 + rs * 16 + hf * 8;
  if (mt < 2) {
    unsigned short* dst = (mt == 0 ? Qb : Kb) + (size_t)b * HW_ * CR_;
    #pragma unroll
    for (int r = 0; r < 8; ++r) {
      size_t row = (size_t)(nbase + r) * CR_;
      dst[row + jcol0] = f2bf(acc0[r] + bias0);
      dst[row + jcol1] = f2bf(acc1[r] + bias1);
    }
  } else {
    unsigned short* dst = Vc + (size_t)b * C_ * HW_;
    const int cbase = (mt - 2) * 64;
    Pack8 p0, p1;
    #pragma unroll
    for (int r = 0; r < 8; ++r) {
      p0.u[r] = f2bf(acc0[r] + bias0);
      p1.u[r] = f2bf(acc1[r] + bias1);
    }
    *(uint4*)(dst + (size_t)(cbase + jcol0) * HW_ + nbase) = p0.q;
    *(uint4*)(dst + (size_t)(cbase + jcol1) * HW_ + nbase) = p1.q;
  }
}

// ---------------------------------------------------------------------------
// Pass 2: flash attention with TDM Q-tile load, double-buffered K/V tiles
// streamed by GLOBAL_LOAD_ASYNC_TO_LDS_B128 (ASYNCcnt-pipelined).
// ---------------------------------------------------------------------------
__global__ __launch_bounds__(256)
void flash_attn_wmma(const unsigned short* __restrict__ Qb,
                     const unsigned short* __restrict__ Kb,
                     const unsigned short* __restrict__ Vc,
                     const float* __restrict__ x,
                     const float* __restrict__ gamma,
                     float* __restrict__ out)
{
  __shared__ __align__(16) unsigned short sQ[64][64];      //  8 KB [n][cr]
  __shared__ __align__(16) unsigned short sK[2][32][64];   //  8 KB [m][cr] x2
  __shared__ __align__(16) unsigned short sV[2][512][32];  // 64 KB [c][m] x2
  __shared__ __align__(16) unsigned short sP[64][32];      //  4 KB [n][m]
  __shared__ float sS[64][33];
  __shared__ float sM[64], sL[64], sCF[64];
  __shared__ float sPmax[4][64], sPsum[4][64];

  const int b   = blockIdx.y;
  const int n0  = blockIdx.x * 64;
  const int tid = threadIdx.x;
  const int wave = tid >> 5;
  const int lane = tid & 31;
  const int hf   = lane >> 4;
  const int l16  = lane & 15;

  const int nsub  = wave & 3;    // O n-subtile
  const int chalf = wave >> 2;   // O c-half
  const int ns    = wave >> 1;   // S n-subtile
  const int ms    = wave & 1;    // S m-subtile

  // ---- Q tile via Tensor Data Mover: 2-D D#, 64x64 bf16 tile ----
  const unsigned long long qg =
      (unsigned long long)(Qb + ((size_t)b * HW_ + n0) * CR_);
  if (wave == 0) {
    v4u g0; v8i g1; v4i g2; v4i g3; v8i g4;
    g0[0] = 1u;                                           // count=1 (valid)
    g0[1] = (unsigned)(unsigned long long)&sQ[0][0];      // lds_addr
    g0[2] = (unsigned)(qg & 0xffffffffu);                 // global_addr lo
    g0[3] = (unsigned)((qg >> 32) & 0x1ffffffu) | (2u << 30); // addr hi | type=2
    g1[0] = (int)(1u << 16);        // data_size=1 (2 bytes/elem)
    g1[1] = (int)(64u << 16);       // tensor_dim0 = 64
    g1[2] = (int)(4096u << 16);     // tensor_dim1 = 4096
    g1[3] = (int)(64u << 16);       // tile_dim0 = 64
    g1[4] = 64;                     // tile_dim1 = 64 rows (tile_dim2 = 0)
    g1[5] = 64;                     // tensor_dim0_stride = 64 (lo32)
    g1[6] = 0;                      // stride hi / dim1_stride lo
    g1[7] = 0;
    g2[0] = g2[1] = g2[2] = g2[3] = 0;
    g3[0] = g3[1] = g3[2] = g3[3] = 0;
    g4[0] = g4[1] = g4[2] = g4[3] = 0;
    g4[4] = g4[5] = g4[6] = g4[7] = 0;
    __builtin_amdgcn_tensor_load_to_lds(g0, g1, g2, g3, g4, 0);
    __builtin_amdgcn_s_wait_tensorcnt(0);
  }
  if (tid < 64) { sM[tid] = -3.0e38f; sL[tid] = 0.0f; }

  v8f acc[16];
  #pragma unroll
  for (int i = 0; i < 16; ++i) acc[i] = (v8f){0,0,0,0,0,0,0,0};

  // ---- async tile streamer: 9 x b128 per thread per (K,V) tile ----
  auto issue_tile = [&](int m0, int buf) {
    {
      int r = tid >> 3, p = (tid & 7) * 8;
      async_b128((unsigned)(unsigned long long)&sK[buf][r][p],
                 (unsigned long long)(Kb + ((size_t)b * HW_ + m0 + r) * CR_ + p));
    }
    #pragma unroll
    for (int ii = 0; ii < 8; ++ii) {
      int i = tid + ii * 256;
      int r = i >> 2, p = (i & 3) * 8;
      async_b128((unsigned)(unsigned long long)&sV[buf][r][p],
                 (unsigned long long)(Vc + ((size_t)b * C_ + r) * HW_ + m0 + p));
    }
  };

  issue_tile(0, 0);                       // prologue: tile 0 -> buffer 0
  __syncthreads();                        // covers sM/sL init + TDM Q visibility

  #pragma unroll 1
  for (int mt = 0; mt < 128; ++mt) {
    const int cur = mt & 1;
    if (mt + 1 < 128) {
      issue_tile((mt + 1) * 32, cur ^ 1);         // stream next tile
      __builtin_prefetch(Vc + ((size_t)b * C_ + tid * 2) * HW_ + (mt + 1) * 32, 0, 1);
      asm volatile("s_wait_asynccnt 0x9" ::: "memory");  // tile mt complete
    } else {
      asm volatile("s_wait_asynccnt 0x0" ::: "memory");
    }
    __syncthreads();

    // ---- S = Q K^T (one 16x16 tile per wave, K-dim = 64) ----
    {
      v8f s = {0,0,0,0,0,0,0,0};
      const unsigned short* qrow = &sQ[ns * 16 + l16][0];
      const unsigned short* krow = &sK[cur][ms * 16 + l16][0];
      #pragma unroll
      for (int ks = 0; ks < 2; ++ks) {
        v16bf a  = frag_a(qrow, ks * 32, hf);
        v16bf bb = frag_b(krow, ks * 32, hf);
        s = __builtin_amdgcn_wmma_f32_16x16x32_bf16(false, a, false, bb, (short)0, s, false, false);
      }
      #pragma unroll
      for (int r = 0; r < 8; ++r)
        sS[ns * 16 + hf * 8 + r][ms * 16 + l16] = s[r];
    }
    __syncthreads();

    // ---- online softmax over the 32 new columns ----
    {
      int r = tid & 63, part = tid >> 6;
      float pm = -3.0e38f;
      #pragma unroll
      for (int j = 0; j < 8; ++j) pm = fmaxf(pm, sS[r][part * 8 + j]);
      sPmax[part][r] = pm;
    }
    __syncthreads();
    if (tid < 64) {
      float mold = sM[tid];
      float mnew = fmaxf(mold, fmaxf(fmaxf(sPmax[0][tid], sPmax[1][tid]),
                                     fmaxf(sPmax[2][tid], sPmax[3][tid])));
      float cf = __expf(mold - mnew);
      sCF[tid] = cf; sM[tid] = mnew; sL[tid] *= cf;
    }
    __syncthreads();
    {
      int r = tid & 63, part = tid >> 6;
      float mnew = sM[r];
      float ps = 0.0f;
      #pragma unroll
      for (int j = 0; j < 8; ++j) {
        float e = __expf(sS[r][part * 8 + j] - mnew);
        sP[r][part * 8 + j] = f2bf(e);
        ps += e;
      }
      sPsum[part][r] = ps;
    }
    __syncthreads();
    if (tid < 64)
      sL[tid] += sPsum[0][tid] + sPsum[1][tid] + sPsum[2][tid] + sPsum[3][tid];

    // ---- rescale accumulators ----
    float cfr[8];
    #pragma unroll
    for (int r = 0; r < 8; ++r) cfr[r] = sCF[nsub * 16 + hf * 8 + r];
    #pragma unroll
    for (int i = 0; i < 16; ++i) {
      #pragma unroll
      for (int r = 0; r < 8; ++r) acc[i][r] *= cfr[r];
    }

    // ---- O += P V^T (16 c-subtiles per wave, K-dim = 32) ----
    {
      v16bf pa = frag_a(&sP[nsub * 16 + l16][0], 0, hf);
      #pragma unroll
      for (int i = 0; i < 16; ++i) {
        v16bf vb = frag_b(&sV[cur][chalf * 256 + i * 16 + l16][0], 0, hf);
        acc[i] = __builtin_amdgcn_wmma_f32_16x16x32_bf16(false, pa, false, vb, (short)0, acc[i], false, false);
      }
    }
    __syncthreads();   // buffer `cur` free for reuse at mt+2
  }

  // ---- epilogue: O/l * gamma + x ----
  float linv[8];
  #pragma unroll
  for (int r = 0; r < 8; ++r) linv[r] = 1.0f / sL[nsub * 16 + hf * 8 + r];
  const float g = gamma[0];
  const int nbase = n0 + nsub * 16 + hf * 8;
  #pragma unroll
  for (int i = 0; i < 16; ++i) {
    const int c = chalf * 256 + i * 16 + l16;
    const size_t base = ((size_t)b * C_ + c) * HW_ + nbase;
    float4 x0 = *(const float4*)(x + base);
    float4 x1 = *(const float4*)(x + base + 4);
    float4 o0, o1;
    o0.x = g * acc[i][0] * linv[0] + x0.x;
    o0.y = g * acc[i][1] * linv[1] + x0.y;
    o0.z = g * acc[i][2] * linv[2] + x0.z;
    o0.w = g * acc[i][3] * linv[3] + x0.w;
    o1.x = g * acc[i][4] * linv[4] + x1.x;
    o1.y = g * acc[i][5] * linv[5] + x1.y;
    o1.z = g * acc[i][6] * linv[6] + x1.z;
    o1.w = g * acc[i][7] * linv[7] + x1.w;
    *(float4*)(out + base)     = o0;
    *(float4*)(out + base + 4) = o1;
  }
}

extern "C" void kernel_launch(void* const* d_in, const int* in_sizes, int n_in,
                              void* d_out, int out_size, void* d_ws, size_t ws_size,
                              hipStream_t stream) {
  (void)in_sizes; (void)n_in; (void)out_size; (void)ws_size;
  const float* x     = (const float*)d_in[0];
  const float* wq    = (const float*)d_in[1];
  const float* bq    = (const float*)d_in[2];
  const float* wk    = (const float*)d_in[3];
  const float* bk    = (const float*)d_in[4];
  const float* wv    = (const float*)d_in[5];
  const float* bv    = (const float*)d_in[6];
  const float* gamma = (const float*)d_in[7];
  float* out = (float*)d_out;

  unsigned short* Qb = (unsigned short*)d_ws;
  unsigned short* Kb = Qb + (size_t)B_ * HW_ * CR_;
  unsigned short* Vc = Kb + (size_t)B_ * HW_ * CR_;

  proj_qkv_wmma<<<dim3(64, 10, B_), 256, 0, stream>>>(x, wq, bq, wk, bk, wv, bv, Qb, Kb, Vc);
  flash_attn_wmma<<<dim3(64, B_), 256, 0, stream>>>(Qb, Kb, Vc, x, gamma, out);
}
